// AttLayer_26800595927501
// MI455X (gfx1250) — matrix operand order repaired and checked by reading, not compiled, not run on hardware
//
#include <hip/hip_runtime.h>

// ---------------------------------------------------------------------------
// CDNA5 (gfx1250, wave32) implementation of the dilated block attention layer.
// B=4, C(Q_DIM)=512, T=8192, E=256, d=256, p=128, W=512, nb=32.
// Strategy: bf16 WMMA (v_wmma_f32_16x16x32_bf16) everywhere, fp32 accum,
// fp32 softmax in LDS, bf16 intermediates in workspace (L2-resident, 192MB L2),
// Tensor Data Mover (tensor_load_to_lds + s_wait_tensorcnt) for input staging.
// ---------------------------------------------------------------------------

typedef __bf16 bf16_t;
typedef bf16_t bf16x16 __attribute__((ext_vector_type(16)));
typedef float  f32x8   __attribute__((ext_vector_type(8)));
typedef unsigned int u32x4 __attribute__((ext_vector_type(4)));
typedef int          i32x4 __attribute__((ext_vector_type(4)));
typedef int          i32x8 __attribute__((ext_vector_type(8)));

#define C_DIM 512
#define T_DIM 8192
#define E_DIM 256
#define D_BLK 256
#define PAD_P 128
#define W_WIN 512
#define SLD   520   // padded LDS row stride for 64-row score / B tiles

// ---- fragment helpers (CDNA5 wave32 WMMA VGPR layouts, CDNA5 ISA) --

// A 16x32 bf16 tile from a row-major matrix (row=M, col=K), lda in elements.
// lanes 0-15: row M=lane, K = 0..15 ; lanes 16-31: row M=lane-16,
// K = 16..31.
__device__ __forceinline__ bf16x16 load_a_rm(const bf16_t* __restrict__ base,
                                             int lda, int lane) {
  int row = lane & 15;
  int kb  = (lane >> 4) << 4;           // 0 or 16
  const bf16_t* p0 = base + row * lda + kb;
  bf16x16 a;
#pragma unroll
  for (int j = 0; j < 16; ++j) { a[j] = p0[j]; }
  return a;
}

// Same as above but source is fp32 (softmax scores in LDS), converted to bf16.
__device__ __forceinline__ bf16x16 load_a_rm_f32(const float* __restrict__ base,
                                                 int lda, int lane) {
  int row = lane & 15;
  int kb  = (lane >> 4) << 4;
  const float* p0 = base + row * lda + kb;
  bf16x16 a;
#pragma unroll
  for (int j = 0; j < 16; ++j) {
    a[j]     = (bf16_t)p0[j];
  }
  return a;
}

// B 32x16 bf16 tile where, per lane, the 16 K-values are contiguous in memory.
// Caller computes p = column_base(col=lane&15) + ((lane>>4)<<4).
__device__ __forceinline__ bf16x16 load_b_kc(const bf16_t* __restrict__ p) {
  bf16x16 b;
#pragma unroll
  for (int j = 0; j < 16; ++j) b[j] = p[j];
  return b;
}

__device__ __forceinline__ f32x8 wmma_bf16(bf16x16 a, bf16x16 b, f32x8 c) {
  // (neg_a, A, neg_b, B, c_mod, C, reuse_a, reuse_b)
  return __builtin_amdgcn_wmma_f32_16x16x32_bf16(false, a, false, b,
                                                 (short)0, c, false, false);
}

// ---------------------------------------------------------------------------
// Kernel 0: fp32 -> bf16 weight conversion into workspace.
// Layout: [Wq 256x512 | Wk | Wv | Wo 512x256], all row-major.
// ---------------------------------------------------------------------------
__global__ void cvt_weights(const float* __restrict__ wq,
                            const float* __restrict__ wk,
                            const float* __restrict__ wv,
                            const float* __restrict__ wo,
                            bf16_t* __restrict__ dst) {
  int idx = blockIdx.x * 256 + threadIdx.x;   // 0 .. 524287
  const float* src; int off;
  if (idx < 131072)      { src = wq; off = idx; }
  else if (idx < 262144) { src = wk; off = idx - 131072; }
  else if (idx < 393216) { src = wv; off = idx - 262144; }
  else                   { src = wo; off = idx - 393216; }
  dst[idx] = (bf16_t)src[off];
}

// ---------------------------------------------------------------------------
// Kernel 1: q/k/v projections.  One WG = (batch b, 64 t-columns).
// GEMM M=256(E) K=512(C) N=64 for each of 3 weights; 192 16x16 tiles / 8 waves.
// Input tile is DMA'd into LDS by the Tensor Data Mover (two 256x64 fp32 D#
// tiles, tracked with TENSORcnt), then converted/transposed to a bf16 tile so
// B-fragments are K-contiguous ds reads.
// Outputs: qT,kT t-major ([t][e]) bf16 ; v e-major ([e][t]) bf16.
// ---------------------------------------------------------------------------
__global__ __launch_bounds__(256) void qkv_kernel(
    const float*  __restrict__ input,
    const bf16_t* __restrict__ wqkv,     // bf16 Wq|Wk|Wv, each 256x512
    const float*  __restrict__ bq, const float* __restrict__ bk,
    const float*  __restrict__ bv,
    bf16_t* __restrict__ qT, bf16_t* __restrict__ kT, bf16_t* __restrict__ v) {
  __shared__ float  Xf[256][64];        // TDM landing buffer (half of K)
  __shared__ bf16_t Xt[64][SLD];        // bf16 B-operand tile, transposed [t][c]
  int tid = threadIdx.x;
  int b   = blockIdx.x >> 7;
  int t0  = (blockIdx.x & 127) << 6;
  int lane = tid & 31, wave = tid >> 5;

  // ---- stage input tile: TDM DMA (fp32) then LDS->LDS transpose/convert ----
#pragma unroll
  for (int half = 0; half < 2; ++half) {
#if __has_builtin(__builtin_amdgcn_tensor_load_to_lds)
    if (wave == 0) {
      const float* gsrc =
          input + ((size_t)b * C_DIM + half * 256) * T_DIM + t0;
      unsigned long long ga = (unsigned long long)(uintptr_t)gsrc;
      unsigned int lds_off  = (unsigned int)(uintptr_t)&Xf[0][0];
      // D# group 0: count=1 | lds_addr | global_addr[56:0] | type=2
      u32x4 g0 = { 1u, lds_off, (unsigned int)ga,
                   (unsigned int)((ga >> 32) & 0x01FFFFFFu) | (2u << 30) };
      // D# group 1: data_size=2 (4B); tensor_dim0=8192; tensor_dim1=256;
      // tile_dim0=64; tile_dim1=256; tensor_dim0_stride=8192
      i32x8 g1 = { (int)(2u << 16),          // wg_mask=0, data_size=4B
                   (int)(8192u << 16),       // tensor_dim0[15:0] @bits63:48
                   (int)(256u << 16),        // tensor_dim1[15:0] @bits95:80
                   (int)(64u << 16),         // tile_dim0 @bits127:112
                   256,                      // tile_dim1 @bits143:128
                   8192,                     // tensor_dim0_stride[31:0]
                   0, 0 };
      i32x4 gz  = { 0, 0, 0, 0 };            // dims 2..4 unused (tile_dim2=0)
      i32x8 gz8 = { 0, 0, 0, 0, 0, 0, 0, 0 };
      // 6-arg toolchain form: (g0, g1, g2, g3, extra, cpol)
      __builtin_amdgcn_tensor_load_to_lds(g0, g1, gz, gz, gz8, 0);
      __builtin_amdgcn_s_wait_tensorcnt(0);
    }
    __syncthreads();
#else
    // fallback: plain coalesced loads
    for (int i = 0; i < 64; ++i) {
      int idx = i * 256 + tid;
      int c = idx >> 6, tc = idx & 63;
      Xf[c][tc] =
          input[((size_t)b * C_DIM + half * 256 + c) * T_DIM + t0 + tc];
    }
    __syncthreads();
#endif
    // transpose + fp32->bf16 into the WMMA B-operand tile
    for (int i = 0; i < 64; ++i) {
      int idx = i * 256 + tid;
      int c = idx >> 6, tc = idx & 63;
      Xt[tc][half * 256 + c] = (bf16_t)Xf[c][tc];
    }
    __syncthreads();
  }

  int col  = lane & 15;
  int kb16 = (lane >> 4) << 4;
  int rb   = (lane >> 4) << 3;

  for (int i = 0; i < 24; ++i) {
    int tt   = wave + (i << 3);          // 0..191
    int wsel = tt >> 6;                  // 0=q 1=k 2=v
    int rem  = tt & 63;
    int m0   = (rem >> 2) << 4;
    int n0   = (rem & 3) << 4;
    const bf16_t* A = wqkv + wsel * 131072 + m0 * C_DIM;
    f32x8 acc = {};
#pragma unroll
    for (int ks = 0; ks < 16; ++ks) {
      bf16x16 af = load_a_rm(A + ks * 32, C_DIM, lane);
      bf16x16 bf = load_b_kc(&Xt[n0 + col][ks * 32 + kb16]);
      acc = wmma_bf16(af, bf, acc);
    }
    const float* bias = (wsel == 0) ? bq : (wsel == 1) ? bk : bv;
    int ncol = t0 + n0 + col;
#pragma unroll
    for (int r = 0; r < 8; ++r) {
      int   m   = m0 + rb + r;
      float val = acc[r] + bias[m];
      if (wsel == 2) {
        v[((size_t)b * E_DIM + m) * T_DIM + ncol] = (bf16_t)val;
      } else {
        bf16_t* dst = (wsel == 0) ? qT : kT;
        dst[((size_t)b * T_DIM + ncol) * E_DIM + m] = (bf16_t)val;
      }
    }
  }
}

// ---------------------------------------------------------------------------
// Kernel 2: block attention. One WG = (batch, block n, 64-query chunk).
// Phase A: S = (Q.K^T)*scale with window/padding mask -> LDS fp32 64x512.
// Phase B: row softmax (wave32 shuffle reductions), in place.
// Phase C: O = att.V^T via WMMA, fused exact GELU, bf16 gT to workspace.
// ---------------------------------------------------------------------------
__global__ __launch_bounds__(256) void attn_kernel(
    const bf16_t* __restrict__ qT, const bf16_t* __restrict__ kT,
    const bf16_t* __restrict__ v,  bf16_t* __restrict__ gT) {
  __shared__ float S[64][SLD];
  int tid = threadIdx.x, lane = tid & 31, wave = tid >> 5;
  int qc    = blockIdx.x & 3;
  int nblk  = (blockIdx.x >> 2) & 31;
  int b     = blockIdx.x >> 7;
  int qbase = nblk * D_BLK + qc * 64;
  int wt0   = nblk * D_BLK - PAD_P;     // window start in global t
  int col   = lane & 15;
  int kb16  = (lane >> 4) << 4;
  int rb    = (lane >> 4) << 3;
  const bf16_t* Qb = qT + ((size_t)b * T_DIM + qbase) * E_DIM;

  // ---- Phase A: energies ----
  for (int i = 0; i < 16; ++i) {
    int tt = wave + (i << 3);            // 0..127
    int m0 = (tt >> 5) << 4;             // query sub-tile
    int n0 = (tt & 31) << 4;             // window sub-tile
    int wt  = wt0 + n0 + col;
    int wtc = min(max(wt, 0), T_DIM - 1);
    const bf16_t* Bb = kT + ((size_t)b * T_DIM + wtc) * E_DIM + kb16;
    __builtin_prefetch(Bb, 0, 1);        // global_prefetch_b8 on the key stream
    f32x8 acc = {};
#pragma unroll
    for (int ks = 0; ks < 8; ++ks) {
      bf16x16 af = load_a_rm(Qb + m0 * E_DIM + ks * 32, E_DIM, lane);
      bf16x16 bf = load_b_kc(Bb + ks * 32);
      acc = wmma_bf16(af, bf, acc);
    }
    // window mask: col < 2d-1 (=511) AND global t within [0,T)
    bool  valid = ((n0 + col) < (2 * D_BLK - 1)) && (wt >= 0) && (wt < T_DIM);
    float sc    = valid ? 0.0625f : 0.0f;   // 1/sqrt(E)
    float addv  = valid ? 0.0f : -1e30f;
#pragma unroll
    for (int r = 0; r < 8; ++r) S[m0 + rb + r][n0 + col] = acc[r] * sc + addv;
  }
  __syncthreads();

  // ---- Phase B: softmax over 512 window columns, 8 rows per wave ----
  for (int r = wave * 8; r < wave * 8 + 8; ++r) {
    float vals[16];
    float mx = -3e30f;
#pragma unroll
    for (int i = 0; i < 16; ++i) {
      vals[i] = S[r][lane + i * 32];
      mx = fmaxf(mx, vals[i]);
    }
#pragma unroll
    for (int off = 16; off > 0; off >>= 1) mx = fmaxf(mx, __shfl_xor(mx, off, 32));
    float sum = 0.f;
#pragma unroll
    for (int i = 0; i < 16; ++i) { vals[i] = __expf(vals[i] - mx); sum += vals[i]; }
#pragma unroll
    for (int off = 16; off > 0; off >>= 1) sum += __shfl_xor(sum, off, 32);
    float inv = 1.0f / sum;
#pragma unroll
    for (int i = 0; i < 16; ++i) S[r][lane + i * 32] = vals[i] * inv;
  }
  __syncthreads();

  // ---- Phase C: O = att . V^T, fused GELU ----
  for (int i = 0; i < 8; ++i) {
    int tt = wave + (i << 3);            // 0..63
    int m0 = (tt >> 4) << 4;             // query sub-tile
    int e0 = (tt & 15) << 4;             // E sub-tile
    const bf16_t* Vb = v + ((size_t)b * E_DIM + e0 + col) * T_DIM;
    f32x8 acc = {};
#pragma unroll
    for (int ks = 0; ks < 16; ++ks) {
      bf16x16 af = load_a_rm_f32(&S[m0][ks * 32], SLD, lane);
      int st = wt0 + ks * 32 + kb16;
      st = min(max(st, 0), T_DIM - 16);  // masked cols have att==0 -> safe clamp
      bf16x16 bf = load_b_kc(Vb + st);
      acc = wmma_bf16(af, bf, acc);
    }
#pragma unroll
    for (int r = 0; r < 8; ++r) {
      float x = acc[r];
      float g = 0.5f * x * (1.0f + erff(x * 0.70710678f));   // exact GELU
      gT[((size_t)b * T_DIM + qbase + m0 + rb + r) * E_DIM + e0 + col] = (bf16_t)g;
    }
  }
}

// ---------------------------------------------------------------------------
// Kernel 3: output projection out = Wo . g + bo, times mask.
// One WG = (batch, 64 t-columns).  M=512 K=256 N=64 -> 128 tiles / 8 waves.
// ---------------------------------------------------------------------------
__global__ __launch_bounds__(256) void outproj_kernel(
    const bf16_t* __restrict__ wo, const float* __restrict__ bo,
    const bf16_t* __restrict__ gT, const float* __restrict__ mask,
    float* __restrict__ out) {
  int tid = threadIdx.x, lane = tid & 31, wave = tid >> 5;
  int b  = blockIdx.x >> 7;
  int t0 = (blockIdx.x & 127) << 6;
  int col  = lane & 15;
  int kb16 = (lane >> 4) << 4;
  int rb   = (lane >> 4) << 3;

  for (int i = 0; i < 16; ++i) {
    int tt = wave + (i << 3);            // 0..127
    int m0 = (tt >> 2) << 4;             // output-channel sub-tile (512/16=32)
    int n0 = (tt & 3) << 4;              // t sub-tile
    const bf16_t* Bb = gT + ((size_t)b * T_DIM + t0 + n0 + col) * E_DIM + kb16;
    f32x8 acc = {};
#pragma unroll
    for (int ks = 0; ks < 8; ++ks) {
      bf16x16 af = load_a_rm(wo + m0 * E_DIM + ks * 32, E_DIM, lane);
      bf16x16 bf = load_b_kc(Bb + ks * 32);
      acc = wmma_bf16(af, bf, acc);
    }
    int   tcol = t0 + n0 + col;
    float mk   = mask[(size_t)b * T_DIM + tcol];
#pragma unroll
    for (int r = 0; r < 8; ++r) {
      int m = m0 + rb + r;
      out[((size_t)b * C_DIM + m) * T_DIM + tcol] = (acc[r] + bo[m]) * mk;
    }
  }
}

// ---------------------------------------------------------------------------
// Host-side launch. Workspace layout (bf16 elements):
//   [0)        wq|wk|wv bf16            393216
//   [393216)   wo bf16                  131072
//   [524288)   qT  (B,T,E) t-major      8388608
//   [+1)       kT  (B,T,E) t-major      8388608
//   [+2)       v   (B,E,T) e-major      8388608
//   [+3)       gT  (B,T,E) t-major      8388608   -> ~65 MB total
// ---------------------------------------------------------------------------
extern "C" void kernel_launch(void* const* d_in, const int* in_sizes, int n_in,
                              void* d_out, int out_size, void* d_ws, size_t ws_size,
                              hipStream_t stream) {
  const float* input = (const float*)d_in[0];
  const float* mask  = (const float*)d_in[1];
  const float* Wq    = (const float*)d_in[2];
  const float* bq    = (const float*)d_in[3];
  const float* Wk    = (const float*)d_in[4];
  const float* bk    = (const float*)d_in[5];
  const float* Wv    = (const float*)d_in[6];
  const float* bv    = (const float*)d_in[7];
  const float* Wo    = (const float*)d_in[8];
  const float* bo    = (const float*)d_in[9];

  bf16_t* ws    = (bf16_t*)d_ws;
  bf16_t* wqkv  = ws;
  bf16_t* wo_bf = ws + 393216;
  bf16_t* qT    = ws + 524288;
  bf16_t* kT    = qT + 8388608;
  bf16_t* vv    = kT + 8388608;
  bf16_t* gT    = vv + 8388608;

  cvt_weights   <<<2048, 256, 0, stream>>>(Wq, Wk, Wv, Wo, ws);
  qkv_kernel    <<<512,  256, 0, stream>>>(input, wqkv, bq, bk, bv, qT, kT, vv);
  attn_kernel   <<<512,  256, 0, stream>>>(qT, kT, vv, gT);
  outproj_kernel<<<512,  256, 0, stream>>>(wo_bf, bo, gT, mask, (float*)d_out);
}
